// Attention_77051713290807
// MI455X (gfx1250) — compile-verified, matrix operand order checked by reading, not run on hardware
//
#include <hip/hip_runtime.h>
#include <hip/hip_bf16.h>

// Problem dims (compile-time constants from the reference)
#define B_  4
#define S_  1024
#define H_  12
#define DM_ 768
#define DH_ 64

typedef __attribute__((ext_vector_type(16))) __bf16        v16bf;
typedef __attribute__((ext_vector_type(8)))  float         v8f;
typedef __attribute__((ext_vector_type(4)))  unsigned int  v4u;

// 16x32 bf16 fragment for one lane (16 bf16 = 8 VGPRs), viewable as two
// 16-byte chunks (ISA layout: lanes 0-15 K-chunks {0..7,16..23},
// lanes 16-31 K-chunks {8..15,24..31} -> per-lane two contiguous 8-elt runs).
union Frag16 {
    v4u            u4[2];
    unsigned short us[16];
    v16bf          bf;
};

__device__ __forceinline__ unsigned short f2bf(float f) {
    unsigned int u = __float_as_uint(f);
    u += 0x7fffu + ((u >> 16) & 1u);   // round-to-nearest-even
    return (unsigned short)(u >> 16);
}

__device__ __forceinline__ v8f wmma_bf16(v16bf a, v16bf b, v8f c) {
    return __builtin_amdgcn_wmma_f32_16x16x32_bf16(
        /*neg_a=*/false, a, /*neg_b=*/false, b,
        /*c_mod=*/(short)0, c, /*reuse_a=*/false, /*reuse_b=*/false);
}

// Load a bf16 fragment whose per-lane row is contiguous in K (global or LDS).
__device__ __forceinline__ Frag16 load_frag(const unsigned short* rowbase,
                                            int k0, int off) {
    Frag16 f;
    f.u4[0] = *(const v4u*)(rowbase + k0 + off);
    f.u4[1] = *(const v4u*)(rowbase + k0 + off + 16);
    return f;
}

// ---------------------------------------------------------------------------
// Async Global->LDS staging (CDNA5 GLOBAL_LOAD_ASYNC_TO_LDS_B128, ASYNCcnt).
// Builtin signature (from hipcc diagnostic): (v4i AS1*, v4i AS3*, Ii, Ii).
// Guarded: falls back to plain b128 copies if the builtin is not exposed.
// ---------------------------------------------------------------------------
#if __has_builtin(__builtin_amdgcn_global_load_async_to_lds_b128)
#define HAS_ASYNC 1
#else
#define HAS_ASYNC 0
#endif

typedef int v4i_t __attribute__((vector_size(4 * sizeof(int))));
typedef __attribute__((address_space(1))) v4i_t glob_v4i;
typedef __attribute__((address_space(3))) v4i_t lds_v4i;

__device__ __forceinline__ void cp16_g2l(const unsigned short* g,
                                         unsigned short* l) {
#if HAS_ASYNC
    __builtin_amdgcn_global_load_async_to_lds_b128(
        (glob_v4i*)(unsigned long long)g, (lds_v4i*)(unsigned long long)l,
        0, 0);
#else
    *(v4u*)l = *(const v4u*)g;
#endif
}

__device__ __forceinline__ void wait_async_pending2() {
#if HAS_ASYNC
#if __has_builtin(__builtin_amdgcn_s_wait_asynccnt)
    __builtin_amdgcn_s_wait_asynccnt(2);
#else
    asm volatile("s_wait_asynccnt 0x2" ::: "memory");
#endif
#endif
}

__device__ __forceinline__ void wait_async_all() {
#if HAS_ASYNC
#if __has_builtin(__builtin_amdgcn_s_wait_asynccnt)
    __builtin_amdgcn_s_wait_asynccnt(0);
#else
    asm volatile("s_wait_asynccnt 0x0" ::: "memory");
#endif
#endif
}

// ---------------------------------------------------------------------------
// Kernel 1: pack weights fp32 -> bf16, transposed so K is contiguous.
//   W_Q/K/V : [h][m(DM)][d(DH)] -> Wt [h][d][m]   (GEMM B with K=DM)
//   W_O     : [h][d(DH)][m(DM)] -> Wt [h][m][d]   (GEMM B with K=DH)
// ---------------------------------------------------------------------------
__global__ __launch_bounds__(256)
void pack_weights(const float* __restrict__ WQ, const float* __restrict__ WK,
                  const float* __restrict__ WV, const float* __restrict__ WO,
                  unsigned short* __restrict__ WQt, unsigned short* __restrict__ WKt,
                  unsigned short* __restrict__ WVt, unsigned short* __restrict__ WOt) {
    int t = blockIdx.x * blockDim.x + threadIdx.x;
    const int n = H_ * DM_ * DH_;
    if (t >= n) return;
    {   // QKV transpose
        int h = t / (DM_ * DH_);
        int rem = t % (DM_ * DH_);
        int m = rem / DH_;
        int d = rem % DH_;
        int o = (h * DH_ + d) * DM_ + m;
        WQt[o] = f2bf(WQ[t]);
        WKt[o] = f2bf(WK[t]);
        WVt[o] = f2bf(WV[t]);
    }
    {   // W_O transpose
        int h = t / (DH_ * DM_);
        int rem = t % (DH_ * DM_);
        int d = rem / DM_;
        int m = rem % DM_;
        WOt[(h * DM_ + m) * DH_ + d] = f2bf(WO[t]);
    }
}

// ---------------------------------------------------------------------------
// Kernel 2: QKV projection.  Per (b,h): [S x DM] @ [DM x DH] (+bias), bf16 WMMA
// with fp32 accumulate.  Block = 8 waves x 16 rows = 128 rows.  N = DH = 64
// = 4 WMMA column tiles.  V stored transposed [d][s] for the PV matmul.
// ---------------------------------------------------------------------------
__global__ __launch_bounds__(256)
void qkv_proj(const float* __restrict__ x,
              const unsigned short* __restrict__ WQt,
              const unsigned short* __restrict__ WKt,
              const unsigned short* __restrict__ WVt,
              const float* __restrict__ bQ, const float* __restrict__ bK,
              const float* __restrict__ bV,
              unsigned short* __restrict__ Qb, unsigned short* __restrict__ Kb,
              unsigned short* __restrict__ Vt) {
    const int lane = threadIdx.x & 31;
    const int wave = threadIdx.x >> 5;
    const int bh = blockIdx.y;
    const int b = bh / H_, h = bh % H_;
    const int m0 = blockIdx.x * 128 + wave * 16;
    const int off = (lane >> 4) * 8;
    const int rowA = m0 + (lane & 15);
    const float* xrow = x + (((size_t)b * S_ + rowA) * H_ + h) * DM_;

    v8f accQ[4], accK[4], accV[4];
#pragma unroll
    for (int t = 0; t < 4; ++t) { v8f z = {}; accQ[t] = z; accK[t] = z; accV[t] = z; }

    for (int k0 = 0; k0 < DM_; k0 += 32) {
        // A fragment: x row, fp32 -> bf16 on the fly (two 8-float runs)
        Frag16 a;
        const float* p0 = xrow + k0 + off;
        if (k0 + 32 < DM_) __builtin_prefetch(xrow + k0 + 32 + off, 0, 0);
#pragma unroll
        for (int i = 0; i < 8; ++i) a.us[i]     = f2bf(p0[i]);
#pragma unroll
        for (int i = 0; i < 8; ++i) a.us[8 + i] = f2bf(p0[16 + i]);

#pragma unroll
        for (int t = 0; t < 4; ++t) {
            const int ncol = t * 16 + (lane & 15);
            const size_t wrow = ((size_t)h * DH_ + ncol) * DM_;
            Frag16 fq = load_frag(WQt + wrow, k0, off);
            accQ[t] = wmma_bf16(a.bf, fq.bf, accQ[t]);
            Frag16 fk = load_frag(WKt + wrow, k0, off);
            accK[t] = wmma_bf16(a.bf, fk.bf, accK[t]);
            Frag16 fv = load_frag(WVt + wrow, k0, off);
            accV[t] = wmma_bf16(a.bf, fv.bf, accV[t]);
        }
    }

    // Epilogue: +bias, convert to bf16, store.  C layout: col = lane&15,
    // row = vgpr + (lane>=16)*8.
#pragma unroll
    for (int t = 0; t < 4; ++t) {
        const int ncol = t * 16 + (lane & 15);
        const float bq = bQ[h * DH_ + ncol];
        const float bk = bK[h * DH_ + ncol];
        const float bv = bV[h * DH_ + ncol];
#pragma unroll
        for (int r = 0; r < 8; ++r) {
            const int row = m0 + r + (lane >> 4) * 8;
            const size_t qi = ((size_t)bh * S_ + row) * DH_ + ncol;
            Qb[qi] = f2bf(accQ[t][r] + bq);
            Kb[qi] = f2bf(accK[t][r] + bk);
            Vt[((size_t)bh * DH_ + ncol) * S_ + row] = f2bf(accV[t][r] + bv);
        }
    }
}

// ---------------------------------------------------------------------------
// Kernel 3: causal flash attention per (b,h).  Block = 128 query rows (8 waves
// x 16 rows).  K/V tiles (32 keys) are staged block-cooperatively into LDS via
// async Global->LDS b128 copies, double-buffered so the next tile's DMA
// overlaps the current tile's WMMAs.  Online softmax in fp32.
// Padded LDS strides: K rows 72, V rows 40 half-words -> the 16-lane b128
// fragment reads start on banks 0,4,...,60 (conflict-free).
// ---------------------------------------------------------------------------
#define KSTR 72
#define VSTR 40
#define LDSW 40
__global__ __launch_bounds__(256)
void flash_attn(const unsigned short* __restrict__ Qb,
                const unsigned short* __restrict__ Kb,
                const unsigned short* __restrict__ Vt,
                unsigned short* __restrict__ Zb) {
    __shared__ unsigned short ldsK[2][32 * KSTR];
    __shared__ unsigned short ldsV[2][64 * VSTR];
    __shared__ unsigned short ldsP[8][16 * LDSW];
    const int tid  = threadIdx.x;
    const int lane = tid & 31;
    const int wave = tid >> 5;
    const int bh = blockIdx.y;
    const int q0 = blockIdx.x * 128 + wave * 16;
    const int off = (lane >> 4) * 8;

    // Per-thread staging coordinates: K tile = 32 rows x 8 chunks of 16B (4KB),
    // V tile = 64 rows x 4 chunks of 16B (4KB); 256 threads x 16B each.
    const int krow = tid >> 3, kchunk = (tid & 7) * 8;
    const int vrow = tid >> 2, vchunk = (tid & 3) * 8;
    const unsigned short* gK = Kb + ((size_t)bh * S_ + krow) * DH_ + kchunk;
    const unsigned short* gV = Vt + ((size_t)bh * DH_ + vrow) * S_ + vchunk;

    // Q fragments (K-dim = DH = 64 -> two 32-wide WMMA steps), loaded once.
    const unsigned short* qrow = Qb + ((size_t)bh * S_ + q0 + (lane & 15)) * DH_;
    Frag16 aQ0 = load_frag(qrow, 0, off);
    Frag16 aQ1 = load_frag(qrow, 32, off);

    v8f accO[4];
#pragma unroll
    for (int t = 0; t < 4; ++t) { v8f z = {}; accO[t] = z; }
    float mrow[8], lrow[8];
#pragma unroll
    for (int r = 0; r < 8; ++r) { mrow[r] = -1e30f; lrow[r] = 0.0f; }

    const int nblk = blockIdx.x * 4 + 4;   // causal key extent of this block

    // Prologue: stage tile 0 into buffer 0.
    cp16_g2l(gK, &ldsK[0][krow * KSTR + kchunk]);
    cp16_g2l(gV, &ldsV[0][vrow * VSTR + vchunk]);

    for (int jb = 0; jb < nblk; ++jb) {
        const int cur = jb & 1;
        const int j0 = jb * 32;
        if (jb + 1 < nblk) {
            // Issue next tile's DMA into the other buffer, then wait for the
            // current tile's 2 older async ops (allow the 2 just issued).
            cp16_g2l(gK + (size_t)(j0 + 32) * DH_,
                     &ldsK[cur ^ 1][krow * KSTR + kchunk]);
            cp16_g2l(gV + (j0 + 32),
                     &ldsV[cur ^ 1][vrow * VSTR + vchunk]);
            wait_async_pending2();
        } else {
            wait_async_all();
        }
        __syncthreads();   // current tile visible to all waves

        // q0 multiple of 16, j0 multiple of 32 => (j0 <= q0+15) <=> (j0 <= q0)
        if (j0 <= q0) {
            // ---- scores S = Q K^T / sqrt(DH), causal mask ----
            float sc[2][8];
#pragma unroll
            for (int t = 0; t < 2; ++t) {
                const unsigned short* kb =
                    &ldsK[cur][(t * 16 + (lane & 15)) * KSTR];
                Frag16 bK0, bK1;
                bK0.u4[0] = *(const v4u*)(kb + off);
                bK0.u4[1] = *(const v4u*)(kb + off + 16);
                bK1.u4[0] = *(const v4u*)(kb + 32 + off);
                bK1.u4[1] = *(const v4u*)(kb + 32 + off + 16);
                v8f c = {};
                c = wmma_bf16(aQ0.bf, bK0.bf, c);
                c = wmma_bf16(aQ1.bf, bK1.bf, c);
#pragma unroll
                for (int r = 0; r < 8; ++r) {
                    const int row = q0 + r + (lane >> 4) * 8;
                    const int col = j0 + t * 16 + (lane & 15);
                    sc[t][r] = (col <= row) ? c[r] * 0.125f : -1e30f;
                }
            }
            // ---- online softmax update (reduce over 16-lane halves) ----
            float alpha[8];
#pragma unroll
            for (int r = 0; r < 8; ++r) {
                float bm = fmaxf(sc[0][r], sc[1][r]);
#pragma unroll
                for (int s = 1; s < 16; s <<= 1)
                    bm = fmaxf(bm, __shfl_xor(bm, s, 32));
                const float mnew = fmaxf(mrow[r], bm);
                alpha[r] = __expf(mrow[r] - mnew);
                sc[0][r] = __expf(sc[0][r] - mnew);
                sc[1][r] = __expf(sc[1][r] - mnew);
                float rs = sc[0][r] + sc[1][r];
#pragma unroll
                for (int s = 1; s < 16; s <<= 1)
                    rs += __shfl_xor(rs, s, 32);
                lrow[r] = lrow[r] * alpha[r] + rs;
                mrow[r] = mnew;
            }
            // ---- P (C-layout) -> per-wave LDS tile -> A-fragment layout ----
#pragma unroll
            for (int t = 0; t < 2; ++t)
#pragma unroll
                for (int r = 0; r < 8; ++r) {
                    const int row = r + (lane >> 4) * 8;
                    const int col = t * 16 + (lane & 15);
                    ldsP[wave][row * LDSW + col] = f2bf(sc[t][r]);
                }
            Frag16 aP;
            {
                const unsigned short* base =
                    (const unsigned short*)__builtin_assume_aligned(
                        &ldsP[wave][(lane & 15) * LDSW + off], 16);
                __builtin_memcpy(&aP.us[0], base, 16);
                __builtin_memcpy(&aP.us[8], base + 16, 16);
            }
            // ---- O = O*alpha + P @ V ----
#pragma unroll
            for (int t = 0; t < 4; ++t) {
#pragma unroll
                for (int r = 0; r < 8; ++r) accO[t][r] *= alpha[r];
                const unsigned short* vb =
                    &ldsV[cur][(t * 16 + (lane & 15)) * VSTR];
                Frag16 bV;
                bV.u4[0] = *(const v4u*)(vb + off);
                bV.u4[1] = *(const v4u*)(vb + off + 16);
                accO[t] = wmma_bf16(aP.bf, bV.bf, accO[t]);
            }
        }
        __syncthreads();   // all waves done reading before buffer reuse
    }

    // ---- normalize and store Z (bf16) ----
#pragma unroll
    for (int t = 0; t < 4; ++t) {
        const int ncol = t * 16 + (lane & 15);
#pragma unroll
        for (int r = 0; r < 8; ++r) {
            const int row = q0 + r + (lane >> 4) * 8;
            Zb[((size_t)bh * S_ + row) * DH_ + ncol] = f2bf(accO[t][r] / lrow[r]);
        }
    }
}

// ---------------------------------------------------------------------------
// Kernel 4: output projection.  Per (b,h): [S x DH] @ [DH x DM] + b_O/H,
// fp32 output.  Grid y tiles DM into 128-wide slabs (8 WMMA column tiles).
// ---------------------------------------------------------------------------
__global__ __launch_bounds__(256)
void out_proj(const unsigned short* __restrict__ Zb,
              const unsigned short* __restrict__ WOt,
              const float* __restrict__ bO, float* __restrict__ out) {
    const int lane = threadIdx.x & 31;
    const int wave = threadIdx.x >> 5;
    const int bh = blockIdx.z;
    const int b = bh / H_, h = bh % H_;
    const int q0 = blockIdx.x * 128 + wave * 16;
    const int n0 = blockIdx.y * 128;
    const int off = (lane >> 4) * 8;

    const unsigned short* zrow = Zb + ((size_t)bh * S_ + q0 + (lane & 15)) * DH_;
    Frag16 a0 = load_frag(zrow, 0, off);
    Frag16 a1 = load_frag(zrow, 32, off);

    v8f acc[8];
#pragma unroll
    for (int t = 0; t < 8; ++t) { v8f z = {}; acc[t] = z; }

#pragma unroll
    for (int t = 0; t < 8; ++t) {
        const int ncol = n0 + t * 16 + (lane & 15);
        const unsigned short* wrow = WOt + ((size_t)h * DM_ + ncol) * DH_;
        Frag16 b0 = load_frag(wrow, 0, off);
        Frag16 b1 = load_frag(wrow, 32, off);
        acc[t] = wmma_bf16(a0.bf, b0.bf, acc[t]);
        acc[t] = wmma_bf16(a1.bf, b1.bf, acc[t]);
    }

#pragma unroll
    for (int t = 0; t < 8; ++t) {
        const int ncol = n0 + t * 16 + (lane & 15);
        const float bias = bO[ncol] * (1.0f / H_);
#pragma unroll
        for (int r = 0; r < 8; ++r) {
            const int row = q0 + r + (lane >> 4) * 8;
            out[(((size_t)b * S_ + row) * H_ + h) * DM_ + ncol] = acc[t][r] + bias;
        }
    }
}

// ---------------------------------------------------------------------------
// Host launcher.  Workspace (bf16): 4 packed weights (4*1.18MB) +
// Q/K/Vt/Z (4*6.29MB)  ~= 29.9 MB total.
// ---------------------------------------------------------------------------
extern "C" void kernel_launch(void* const* d_in, const int* in_sizes, int n_in,
                              void* d_out, int out_size, void* d_ws, size_t ws_size,
                              hipStream_t stream) {
    const float* x  = (const float*)d_in[0];  // [B,S,H,DM]
    const float* WQ = (const float*)d_in[1];  // [H,DM,DH]
    const float* bQ = (const float*)d_in[2];  // [H,DH]
    const float* WK = (const float*)d_in[3];
    const float* bK = (const float*)d_in[4];
    const float* WV = (const float*)d_in[5];
    const float* bV = (const float*)d_in[6];
    const float* WO = (const float*)d_in[7];  // [H,DH,DM]
    const float* bO = (const float*)d_in[8];  // [DM]
    float* out = (float*)d_out;               // [B,S,H,DM] fp32

    const size_t szW = (size_t)H_ * DM_ * DH_;       // packed weight elements
    const size_t szQ = (size_t)B_ * H_ * S_ * DH_;   // per-tensor QKV elements
    unsigned short* WQt = (unsigned short*)d_ws;
    unsigned short* WKt = WQt + szW;
    unsigned short* WVt = WKt + szW;
    unsigned short* WOt = WVt + szW;
    unsigned short* Qb  = WOt + szW;
    unsigned short* Kb  = Qb + szQ;
    unsigned short* Vt  = Kb + szQ;
    unsigned short* Zb  = Vt + szQ;

    const int npack = H_ * DM_ * DH_;
    pack_weights<<<(npack + 255) / 256, 256, 0, stream>>>(WQ, WK, WV, WO,
                                                          WQt, WKt, WVt, WOt);
    dim3 gq(S_ / 128, B_ * H_);
    qkv_proj<<<gq, 256, 0, stream>>>(x, WQt, WKt, WVt, bQ, bK, bV, Qb, Kb, Vt);
    flash_attn<<<gq, 256, 0, stream>>>(Qb, Kb, Vt, Zb);
    dim3 go(S_ / 128, DM_ / 128, B_ * H_);
    out_proj<<<go, 256, 0, stream>>>(Zb, WOt, bO, out);
}
